// moe_7404523618953
// MI455X (gfx1250) — matrix operand order repaired and checked by reading, not compiled, not run on hardware
//
#include <hip/hip_runtime.h>
#include <cstddef>

#define D_MODEL 512
#define D_HIDDEN 1024
#define GU2 2048            // 2 * D_HIDDEN
#define N_EXPERT 8
#define TILE_M 32           // tokens per workgroup (2 subtiles of 16)
#define SUB 2               // 16-row subtiles per workgroup

#define NT_GU (GU2 / 16)        // 128 column tiles in gate/up weights
#define KT_GU (D_MODEL / 32)    // 16 k tiles
#define NT_D  (D_MODEL / 16)    // 32 column tiles in down weights
#define KT_D  (D_HIDDEN / 32)   // 32 k tiles

typedef __attribute__((ext_vector_type(16))) __bf16        v16bf;
typedef __attribute__((ext_vector_type(8)))  float         v8f;
typedef __attribute__((ext_vector_type(4)))  unsigned int  v4u;
typedef __attribute__((ext_vector_type(8)))  int           v8i;
typedef __attribute__((ext_vector_type(4)))  int           v4i;

// ---------------------------------------------------------------------------
// Gating: logits = x @ Wg, top-2, softmax over the two, scatter to comb[N][E]
// ---------------------------------------------------------------------------
__global__ void moe_gate_kernel(const float* __restrict__ x,
                                const float* __restrict__ Wg,
                                float* __restrict__ comb, int N)
{
    int n = blockIdx.x * blockDim.x + threadIdx.x;
    if (n >= N) return;
    float acc[N_EXPERT];
#pragma unroll
    for (int e = 0; e < N_EXPERT; ++e) acc[e] = 0.f;
    const float* xr = x + (size_t)n * D_MODEL;
    for (int d = 0; d < D_MODEL; ++d) {
        float xv = xr[d];
#pragma unroll
        for (int e = 0; e < N_EXPERT; ++e) acc[e] += xv * Wg[d * N_EXPERT + e];
    }
    int   i1 = 0; float l1 = acc[0];
#pragma unroll
    for (int e = 1; e < N_EXPERT; ++e) if (acc[e] > l1) { l1 = acc[e]; i1 = e; }
    int   i2 = -1; float l2 = -__builtin_inff();
#pragma unroll
    for (int e = 0; e < N_EXPERT; ++e) if (e != i1 && acc[e] > l2) { l2 = acc[e]; i2 = e; }
    float w2 = 1.f / (1.f + __expf(l1 - l2));
    float w1 = 1.f - w2;
#pragma unroll
    for (int e = 0; e < N_EXPERT; ++e) comb[n * N_EXPERT + e] = 0.f;
    comb[n * N_EXPERT + i1] = w1;
    comb[n * N_EXPERT + i2] = w2;
}

// ---------------------------------------------------------------------------
// Weight pre-swizzle: f32 row-major [K x N] -> bf16 WMMA B-fragment order.
// Packed layout: [e][ntile][ktile][lane(32)][elem(16)] bf16, 32 B per lane.
// B-fragment element i of lane l = W[ kt*32 + (l>>4)*16 + i ][ nt*16 + (l&15) ]
// ---------------------------------------------------------------------------
__global__ void moe_pack_kernel(const float* __restrict__ Wgu,
                                const float* __restrict__ Wd,
                                __bf16* __restrict__ pgu,
                                __bf16* __restrict__ pwd)
{
    size_t idx = (size_t)blockIdx.x * blockDim.x + threadIdx.x;
    const size_t NGU = (size_t)N_EXPERT * NT_GU * KT_GU * 32 * 16;
    const size_t NWD = (size_t)N_EXPERT * NT_D  * KT_D  * 32 * 16;
    if (idx < NGU) {
        int i    = idx & 15;
        int lane = (idx >> 4) & 31;
        size_t r = idx >> 9;
        int kt = r % KT_GU;  r /= KT_GU;
        int nt = r % NT_GU;  r /= NT_GU;
        int e  = (int)r;
        int k   = kt * 32 + ((lane >> 4) << 4) + i;
        int col = nt * 16 + (lane & 15);
        pgu[idx] = (__bf16)Wgu[((size_t)e * D_MODEL + k) * GU2 + col];
    }
    if (idx < NWD) {
        int i    = idx & 15;
        int lane = (idx >> 4) & 31;
        size_t r = idx >> 9;
        int kt = r % KT_D;   r /= KT_D;
        int nt = r % NT_D;   r /= NT_D;
        int e  = (int)r;
        int k   = kt * 32 + ((lane >> 4) << 4) + i;
        int col = nt * 16 + (lane & 15);
        pwd[idx] = (__bf16)Wd[((size_t)e * D_HIDDEN + k) * D_MODEL + col];
    }
}

// ---------------------------------------------------------------------------
// WMMA helpers (wave32, CDNA5 ISA 7.12.2 layouts)
// ---------------------------------------------------------------------------
__device__ inline v8f wmma_bf16(v16bf a, v16bf b, v8f c)
{
    return __builtin_amdgcn_wmma_f32_16x16x32_bf16(
        false, a, false, b, (short)0, c, false, false);
}

// B fragment from f32 row-major global (fallback path, converts in-register).
__device__ inline v16bf load_b_f32(const float* __restrict__ W,
                                   int ncols, int kbase, int colbase, int lane)
{
    int n = lane & 15;
    int khalf = lane >> 4;
    v16bf b;
#pragma unroll
    for (int v = 0; v < 8; ++v) {
        int k = kbase + (khalf << 4) + (v << 1);
        const float* p = W + (size_t)k * ncols + colbase + n;
        b[2 * v]     = (__bf16)p[0];
        b[2 * v + 1] = (__bf16)p[ncols];
    }
    return b;
}

// ---------------------------------------------------------------------------
// Expert kernel: grid = (token tiles of 32) x (experts). Each workgroup does
// one expert's GEMM1 -> SwiGLU -> GEMM2 for a 32-token tile and atomically
// adds comb-weighted results to the (pre-zeroed) output.
// Two 16-row subtiles share every B fragment -> 2 WMMA per 32-byte B load.
// X staged by the Tensor Data Mover, repacked to fragment-ordered bf16 A.
// ---------------------------------------------------------------------------
union StageU {
    float  stage[TILE_M * D_MODEL];          // 64 KB: TDM landing pad for X
    __bf16 hf[SUB][KT_D * 32 * 16];          // 2 x 32 KB: fragment-ordered H
};

template <bool PACKED>
__global__ void __launch_bounds__(256)
moe_expert_kernel(const float* __restrict__ x,
                  const float* __restrict__ Wgu,
                  const float* __restrict__ bgu,
                  const float* __restrict__ Wd,
                  const float* __restrict__ bd,
                  const float* __restrict__ comb,
                  const __bf16* __restrict__ pgu,
                  const __bf16* __restrict__ pwd,
                  float* __restrict__ out)
{
    __shared__ __align__(32) StageU sU;
    __shared__ __align__(32) __bf16 sXf[SUB][KT_GU * 32 * 16];  // 2 x 16 KB
    __shared__ float sComb[TILE_M];

    const int row0  = blockIdx.x * TILE_M;
    const int e     = blockIdx.y;
    const int tid   = threadIdx.x;
    const int lane  = tid & 31;
    const int wave  = tid >> 5;
    const int nlo   = lane & 15;
    const int khalf = lane >> 4;

    // ---- Stage X tile (32 x 512 f32, contiguous) via Tensor Data Mover. ----
    if (wave == 0) {
        unsigned lds_off = (unsigned)(size_t)(&sU.stage[0]);
        unsigned long long ga =
            (unsigned long long)(const void*)(x + (size_t)row0 * D_MODEL);
        const unsigned elems = TILE_M * D_MODEL;             // 16384 f32
        // D# group0: count=1 | lds_addr | global_addr[56:0] | type=2
        v4u g0 = { 1u,
                   lds_off,
                   (unsigned)(ga & 0xffffffffu),
                   (unsigned)((ga >> 32) & 0x01ffffffu) | 0x80000000u };
        // D# group1: data_size=4B; tensor_dim0=elems; tensor_dim1=1;
        //            tile_dim0=elems; tile_dim1=1; dim0_stride=elems
        v8i g1 = { (int)(2u << 16),
                   (int)(elems << 16),
                   (int)((elems >> 16) | (1u << 16)),
                   (int)(elems << 16),
                   (int)1,
                   (int)elems,
                   0, 0 };
        v4i z4 = { 0, 0, 0, 0 };
        v8i z8 = { 0, 0, 0, 0, 0, 0, 0, 0 };
        __builtin_amdgcn_tensor_load_to_lds(g0, g1, z4, z4, z8, 0);
        __builtin_amdgcn_s_wait_tensorcnt(0);
    }
    if (tid < TILE_M)
        sComb[tid] = comb[(row0 + tid) * N_EXPERT + e];
    __syncthreads();

    // ---- Repack staged f32 X into fragment-ordered bf16 A tiles. ----
    // A elem i of lane l at ktile kt: k = kt*32 + 16*(l>>4) + i,
    // row m = subtile*16 + (l & 15).
    for (int i = tid; i < SUB * KT_GU * 32 * 16; i += 256) {
        int s  = i >> 13;                 // 8192 fragment elems per subtile
        int r  = i & 8191;
        int ei = r & 15;
        int ln = (r >> 4) & 31;
        int kt = r >> 9;
        int k  = kt * 32 + ((ln >> 4) << 4) + ei;
        int m  = (s << 4) + (ln & 15);
        sXf[s][r] = (__bf16)sU.stage[m * D_MODEL + k];
    }
    __syncthreads();

    // Pull this expert's down-proj weights toward L2 while GEMM1 runs.
    if (PACKED)
        __builtin_prefetch(pwd + ((size_t)e * NT_D * KT_D * 512) +
                           (size_t)tid * 64, 0, 1);
    else
        __builtin_prefetch(Wd + (size_t)e * D_HIDDEN * D_MODEL +
                           (size_t)tid * 64, 0, 1);

    // ---- GEMM1 + SwiGLU -> fragment-ordered bf16 H (both subtiles). ----
    for (int jt = wave * 8; jt < wave * 8 + 8; ++jt) {
        const int gcol = jt * 16;
        v8f cg[SUB], cu[SUB];
#pragma unroll
        for (int s = 0; s < SUB; ++s) { cg[s] = (v8f){}; cu[s] = (v8f){}; }
#pragma unroll 4
        for (int kt = 0; kt < KT_GU; ++kt) {
            v16bf bg, bu;
            if (PACKED) {
                bg = *(const v16bf*)(pgu +
                    ((((size_t)e * NT_GU + jt) * KT_GU + kt) * 32 + lane) * 16);
                bu = *(const v16bf*)(pgu +
                    ((((size_t)e * NT_GU + 64 + jt) * KT_GU + kt) * 32 + lane) * 16);
            } else {
                const float* We = Wgu + (size_t)e * D_MODEL * GU2;
                bg = load_b_f32(We, GU2, kt * 32, gcol, lane);
                bu = load_b_f32(We, GU2, kt * 32, D_HIDDEN + gcol, lane);
            }
#pragma unroll
            for (int s = 0; s < SUB; ++s) {
                v16bf a = *(const v16bf*)&sXf[s][(kt * 32 + lane) * 16];
                cg[s] = wmma_bf16(a, bg, cg[s]);
                cu[s] = wmma_bf16(a, bu, cu[s]);
            }
        }
        const float bg_ = bgu[e * GU2 + gcol + nlo];
        const float bu_ = bgu[e * GU2 + D_HIDDEN + gcol + nlo];
        // Scatter SwiGLU into fragment-ordered H; col identical for all 8
        // elements of a fragment, only row m varies.
        const int col = gcol + nlo;
        const int kt2 = col >> 5;
        const int kk2 = col & 31;
        const int hb  = kk2 >> 4;
        const int i2  = kk2 & 15;
#pragma unroll
        for (int s = 0; s < SUB; ++s) {
#pragma unroll
            for (int v = 0; v < 8; ++v) {
                const int m = v + (khalf << 3);
                float g = cg[s][v] + bg_;
                float u = cu[s][v] + bu_;
                float h = (g / (1.f + __expf(-g))) * u;   // SiLU(g) * u
                sU.hf[s][(kt2 * 32 + hb * 16 + m) * 16 + i2] = (__bf16)h;
            }
        }
    }
    __syncthreads();   // H fully written before GEMM2 reads all of it

    // ---- GEMM2 + combine-weighted atomic accumulation into out. ----
    for (int j = 0; j < 4; ++j) {
        const int nt = wave * 4 + j;
        const int colbase = nt * 16;
        v8f y[SUB];
#pragma unroll
        for (int s = 0; s < SUB; ++s) y[s] = (v8f){};
#pragma unroll 4
        for (int kt = 0; kt < KT_D; ++kt) {
            v16bf b;
            if (PACKED) {
                b = *(const v16bf*)(pwd +
                    ((((size_t)e * NT_D + nt) * KT_D + kt) * 32 + lane) * 16);
            } else {
                b = load_b_f32(Wd + (size_t)e * D_HIDDEN * D_MODEL,
                               D_MODEL, kt * 32, colbase, lane);
            }
#pragma unroll
            for (int s = 0; s < SUB; ++s) {
                v16bf a = *(const v16bf*)&sU.hf[s][(kt * 32 + lane) * 16];
                y[s] = wmma_bf16(a, b, y[s]);
            }
        }
        const float bd_ = bd[e * D_MODEL + colbase + nlo];
#pragma unroll
        for (int s = 0; s < SUB; ++s) {
#pragma unroll
            for (int v = 0; v < 8; ++v) {
                const int m = v + (khalf << 3);
                const int row = (s << 4) + m;
                float val = sComb[row] * (y[s][v] + bd_);
                unsafeAtomicAdd(&out[(size_t)(row0 + row) * D_MODEL +
                                     colbase + nlo], val);
            }
        }
    }
}

// ---------------------------------------------------------------------------
extern "C" void kernel_launch(void* const* d_in, const int* in_sizes, int n_in,
                              void* d_out, int out_size, void* d_ws, size_t ws_size,
                              hipStream_t stream)
{
    const float* x   = (const float*)d_in[0];
    const float* Wg  = (const float*)d_in[1];
    const float* Wgu = (const float*)d_in[2];
    const float* bgu = (const float*)d_in[3];
    const float* Wd  = (const float*)d_in[4];
    const float* bd  = (const float*)d_in[5];
    float* out = (float*)d_out;

    const int N = in_sizes[0] / D_MODEL;                  // 2048 tokens

    const size_t comb_bytes = (size_t)N * N_EXPERT * sizeof(float);
    const size_t ngu = (size_t)N_EXPERT * NT_GU * KT_GU * 32 * 16;  // 8.39M
    const size_t nwd = (size_t)N_EXPERT * NT_D  * KT_D  * 32 * 16;  // 4.19M
    const size_t comb_pad = (comb_bytes + 255) & ~(size_t)255;
    const size_t need = comb_pad + (ngu + nwd) * 2;

    float* comb = (float*)d_ws;
    moe_gate_kernel<<<(N + 255) / 256, 256, 0, stream>>>(x, Wg, comb, N);

    // Output accumulated via atomics -> zero it first (capture-safe).
    hipMemsetAsync(out, 0, (size_t)out_size * sizeof(float), stream);

    dim3 grid(N / TILE_M, N_EXPERT);
    if (ws_size >= need) {
        __bf16* pgu = (__bf16*)((char*)d_ws + comb_pad);
        __bf16* pwd = pgu + ngu;
        const size_t npack = ngu > nwd ? ngu : nwd;
        moe_pack_kernel<<<(int)((npack + 255) / 256), 256, 0, stream>>>(
            Wgu, Wd, pgu, pwd);
        moe_expert_kernel<true><<<grid, 256, 0, stream>>>(
            x, Wgu, bgu, Wd, bd, comb, pgu, pwd, out);
    } else {
        moe_expert_kernel<false><<<grid, 256, 0, stream>>>(
            x, Wgu, bgu, Wd, bd, comb, nullptr, nullptr, out);
    }
}
